// StyleAggregation_17978733101383
// MI455X (gfx1250) — matrix operand order repaired and checked by reading, not compiled, not run on hardware
//
#include <hip/hip_runtime.h>

// ---------------------------------------------------------------------------
// StyleAggregation (Performer linear attention) for MI455X / gfx1250
// All four GEMMs run on v_wmma_f32_16x16x32_f16 (wave32).
// ---------------------------------------------------------------------------

typedef _Float16 v16h __attribute__((ext_vector_type(16)));
typedef _Float16 v8h  __attribute__((ext_vector_type(8)));
typedef _Float16 v4h  __attribute__((ext_vector_type(4)));
typedef float    v8f  __attribute__((ext_vector_type(8)));
typedef float    v4f  __attribute__((ext_vector_type(4)));

union H16 { v16h v; v8h h[2]; };

static constexpr int NB = 8;     // batches
static constexpr int LQ = 2048;  // query length
static constexpr int LK = 8192;  // K*L style length
static constexpr int DD = 512;   // feature dim d
static constexpr int NM = 256;   // random features m

static constexpr float PROJ_SCALE = 0.21022410381342863f;  // 512^-0.25 (folded into W)
static constexpr float SQ_SCALE   = 0.022097086912079608f; // 1/(2*sqrt(512))
static constexpr float M_SCALE    = 0.0625f;               // 256^-0.5

__device__ __forceinline__ v8f wmma16(v16h a, v16h b, v8f c) {
  return __builtin_amdgcn_wmma_f32_16x16x32_f16(false, a, false, b, (short)0, c,
                                                false, false);
}

// ---------------------------------------------------------------------------
// Kernel 0: W [512][256] fp32 -> W16T [m=256][d=512] f16, scaled by d^-0.25
// ---------------------------------------------------------------------------
__global__ __launch_bounds__(256) void k_wcvt(const float* __restrict__ W,
                                              _Float16* __restrict__ W16T) {
  int idx = blockIdx.x * 256 + threadIdx.x;     // 131072 elements
  int m = idx >> 9;
  int d = idx & 511;
  W16T[idx] = (_Float16)(W[(size_t)d * NM + m] * PROJ_SCALE);
}

// ---------------------------------------------------------------------------
// Kernel 1: phi_k.  proj[kl, m] = style * Wscaled  (M=kl tile 64, N=256 all m)
// Stores phiKT [b][m][8192 kl] (transposed, per-lane contiguous 16B).
// Style fp32 staged through LDS as f16; ||x||^2 fused into staging.
// ---------------------------------------------------------------------------
__global__ __launch_bounds__(256) void k_phik(const float* __restrict__ style,
                                              const _Float16* __restrict__ W16T,
                                              _Float16* __restrict__ phiKT) {
  __shared__ __align__(16) _Float16 ldsA[64][32];
  __shared__ float ldsSqP[64][4];

  const int tid = threadIdx.x, lane = tid & 31, wave = tid >> 5;
  const int waveM = wave & 1;   // 2 groups x 32 kl
  const int waveN = wave >> 1;  // 4 groups x 64 m
  const int b = blockIdx.x >> 7;
  const int klT = (blockIdx.x & 127) << 6;

  v8f acc[2][4];
  const v8f vzero = {0.f, 0.f, 0.f, 0.f, 0.f, 0.f, 0.f, 0.f};
#pragma unroll
  for (int i = 0; i < 2; ++i)
#pragma unroll
    for (int j = 0; j < 4; ++j) acc[i][j] = vzero;

  const int srow = tid >> 2, sseg = (tid & 3) << 3;
  const float* srcRow = style + ((size_t)b * LK + klT + srow) * DD + sseg;
  float sq = 0.f;

  const int koff8  = (lane >> 4) << 3;
  const int koff16 = (lane >> 4) << 4;

  for (int kt = 0; kt < 16; ++kt) {
    const int kb = kt << 5;
    // ---- stage 64x32 fp32 tile -> f16 LDS, accumulate sum of squares ----
    v4f x0 = *(const v4f*)(srcRow + kb);
    v4f x1 = *(const v4f*)(srcRow + kb + 4);
    if (kt < 15) __builtin_prefetch(srcRow + kb + 32, 0, 1);
    v8h hv;
#pragma unroll
    for (int j = 0; j < 4; ++j) {
      sq += x0[j] * x0[j] + x1[j] * x1[j];
      hv[j] = (_Float16)x0[j];
      hv[j + 4] = (_Float16)x1[j];
    }
    *(v8h*)&ldsA[srow][sseg] = hv;
    __syncthreads();

    H16 af[2], bf[4];
#pragma unroll
    for (int mf = 0; mf < 2; ++mf) {
      const int row = (waveM << 5) + (mf << 4) + (lane & 15);
      af[mf].h[0] = *(const v8h*)&ldsA[row][koff8];
      af[mf].h[1] = *(const v8h*)&ldsA[row][16 + koff8];
    }
#pragma unroll
    for (int nf = 0; nf < 4; ++nf) {
      const int m = (waveN << 6) + (nf << 4) + (lane & 15);
      const _Float16* wp = W16T + (size_t)m * DD + kb + koff8;
      bf[nf].h[0] = *(const v8h*)wp;
      bf[nf].h[1] = *(const v8h*)(wp + 16);
    }
#pragma unroll
    for (int mf = 0; mf < 2; ++mf)
#pragma unroll
      for (int nf = 0; nf < 4; ++nf)
        acc[mf][nf] = wmma16(af[mf].v, bf[nf].v, acc[mf][nf]);
    __syncthreads();
  }

  ldsSqP[srow][tid & 3] = sq;   // deterministic partials, fixed-order reduce
  __syncthreads();

#pragma unroll
  for (int mf = 0; mf < 2; ++mf) {
    const int rb = (waveM << 5) + (mf << 4) + ((lane >> 4) << 3);
    float sqv[8];
#pragma unroll
    for (int r = 0; r < 8; ++r) {
      const float* p = ldsSqP[rb + r];
      sqv[r] = ((p[0] + p[1]) + (p[2] + p[3])) * SQ_SCALE;
    }
#pragma unroll
    for (int nf = 0; nf < 4; ++nf) {
      const int mcol = (waveN << 6) + (nf << 4) + (lane & 15);
      v8h hv;
#pragma unroll
      for (int r = 0; r < 8; ++r)
        hv[r] = (_Float16)(__expf(acc[mf][nf][r] - sqv[r]) * M_SCALE);
      _Float16* dst = phiKT + ((size_t)b * NM + mcol) * LK + klT + rb;
      *(v8h*)dst = hv;   // 16B contiguous along kl for fixed m (transposed!)
    }
  }
}

// ---------------------------------------------------------------------------
// Kernel 2: Z[b][m] = sum over kl of phiKT row (deterministic tree-free sum)
// ---------------------------------------------------------------------------
__global__ __launch_bounds__(256) void k_zsum(const _Float16* __restrict__ phiKT,
                                              float* __restrict__ Z) {
  __shared__ float part[256];
  const int tid = threadIdx.x, lane = tid & 31, wave = tid >> 5;
  const int row = blockIdx.x * 8 + wave;       // row = b*256 + m
  const _Float16* p = phiKT + (size_t)row * LK;
  float s = 0.f;
  for (int k = 0; k < 32; ++k) {
    v8h h = *(const v8h*)(p + ((size_t)(k * 32 + lane) << 3));
#pragma unroll
    for (int e = 0; e < 8; ++e) s += (float)h[e];
  }
  part[tid] = s;
  __syncthreads();
  if (lane == 0) {
    float t = 0.f;
    for (int i = 0; i < 32; ++i) t += part[(wave << 5) + i];
    Z[row] = t;
  }
}

// ---------------------------------------------------------------------------
// Kernel 3: phi_q, computed transposed: projT[m, l] = Wscaled^T-GEMM.
// A = W16T rows (f16 global), B = content rows staged through LDS.
// Stores phiQ [b][l][m] row-major (per-lane contiguous 16B along m).
// ---------------------------------------------------------------------------
__global__ __launch_bounds__(256) void k_phiq(const float* __restrict__ content,
                                              const _Float16* __restrict__ W16T,
                                              _Float16* __restrict__ phiQ) {
  __shared__ __align__(16) _Float16 ldsB[64][32];
  __shared__ float ldsSqP[64][4];

  const int tid = threadIdx.x, lane = tid & 31, wave = tid >> 5;
  const int waveM = wave & 3;   // 4 groups x 64 m
  const int waveN = wave >> 2;  // 2 groups x 32 l
  const int b = blockIdx.x >> 5;
  const int lT = (blockIdx.x & 31) << 6;

  v8f acc[4][2];
  const v8f vzero = {0.f, 0.f, 0.f, 0.f, 0.f, 0.f, 0.f, 0.f};
#pragma unroll
  for (int i = 0; i < 4; ++i)
#pragma unroll
    for (int j = 0; j < 2; ++j) acc[i][j] = vzero;

  const int srow = tid >> 2, sseg = (tid & 3) << 3;
  const float* srcRow = content + ((size_t)b * LQ + lT + srow) * DD + sseg;
  float sq = 0.f;

  const int koff8  = (lane >> 4) << 3;
  const int koff16 = (lane >> 4) << 4;

  for (int kt = 0; kt < 16; ++kt) {
    const int kb = kt << 5;
    v4f x0 = *(const v4f*)(srcRow + kb);
    v4f x1 = *(const v4f*)(srcRow + kb + 4);
    v8h hv;
#pragma unroll
    for (int j = 0; j < 4; ++j) {
      sq += x0[j] * x0[j] + x1[j] * x1[j];
      hv[j] = (_Float16)x0[j];
      hv[j + 4] = (_Float16)x1[j];
    }
    *(v8h*)&ldsB[srow][sseg] = hv;
    __syncthreads();

    H16 af[4], bf[2];
#pragma unroll
    for (int mf = 0; mf < 4; ++mf) {
      const int mrow = (waveM << 6) + (mf << 4) + (lane & 15);
      const _Float16* wp = W16T + (size_t)mrow * DD + kb + koff8;
      af[mf].h[0] = *(const v8h*)wp;
      af[mf].h[1] = *(const v8h*)(wp + 16);
    }
#pragma unroll
    for (int nf = 0; nf < 2; ++nf) {
      const int lrow = (waveN << 5) + (nf << 4) + (lane & 15);
      bf[nf].h[0] = *(const v8h*)&ldsB[lrow][koff8];
      bf[nf].h[1] = *(const v8h*)&ldsB[lrow][koff8 + 16];
    }
#pragma unroll
    for (int mf = 0; mf < 4; ++mf)
#pragma unroll
      for (int nf = 0; nf < 2; ++nf)
        acc[mf][nf] = wmma16(af[mf].v, bf[nf].v, acc[mf][nf]);
    __syncthreads();
  }

  ldsSqP[srow][tid & 3] = sq;
  __syncthreads();

#pragma unroll
  for (int nf = 0; nf < 2; ++nf) {
    const int ll = (waveN << 5) + (nf << 4) + (lane & 15);
    const float* p = ldsSqP[ll];
    const float sqv = ((p[0] + p[1]) + (p[2] + p[3])) * SQ_SCALE;
#pragma unroll
    for (int mf = 0; mf < 4; ++mf) {
      const int mb = (waveM << 6) + (mf << 4) + ((lane >> 4) << 3);
      v8h hv;
#pragma unroll
      for (int r = 0; r < 8; ++r)
        hv[r] = (_Float16)(__expf(acc[mf][nf][r] - sqv) * M_SCALE);
      _Float16* dst = phiQ + ((size_t)b * LQ + lT + ll) * NM + mb;
      *(v8h*)dst = hv;   // 16B contiguous along m for fixed l
    }
  }
}

// ---------------------------------------------------------------------------
// Kernel 4: S-GEMM. S[mq,d] = sum_kl phiK[kl,mq]*style[kl,d], K-split by 4.
// A = phiKT rows (f16), B = style tile transposed to f16 through LDS.
// Stores fp32 partials S32p[b][ks][d][mq] (per-lane contiguous 32B).
// ---------------------------------------------------------------------------
__global__ __launch_bounds__(256) void k_sgemm(const float* __restrict__ style,
                                               const _Float16* __restrict__ phiKT,
                                               float* __restrict__ S32p) {
  __shared__ __align__(16) _Float16 ldsBT[64][32];  // [d_local][kl_local]

  const int tid = threadIdx.x, lane = tid & 31, wave = tid >> 5;
  const int waveM = wave & 3;   // 4 groups x 64 mq
  const int waveN = wave >> 2;  // 2 groups x 32 d
  const int b = blockIdx.x >> 5;
  const int rest = blockIdx.x & 31;
  const int dT = (rest & 7) << 6;   // 8 d-tiles of 64
  const int ks = rest >> 3;         // 4-way K split
  const int kl0 = ks << 11;         // 2048 kl per split

  v8f acc[4][2];
  const v8f vzero = {0.f, 0.f, 0.f, 0.f, 0.f, 0.f, 0.f, 0.f};
#pragma unroll
  for (int i = 0; i < 4; ++i)
#pragma unroll
    for (int j = 0; j < 2; ++j) acc[i][j] = vzero;

  const int klrow = tid >> 3, dseg = (tid & 7) << 3;
  const int koff8  = (lane >> 4) << 3;
  const int koff16 = (lane >> 4) << 4;

  for (int kt = 0; kt < 64; ++kt) {
    const int kb = kl0 + (kt << 5);
    // ---- stage style [32 kl][64 d] fp32 -> transposed f16 LDS [d][kl] ----
    const float* sp = style + ((size_t)b * LK + kb + klrow) * DD + dT + dseg;
    v4f x0 = *(const v4f*)sp;
    v4f x1 = *(const v4f*)(sp + 4);
    if (kt < 63) __builtin_prefetch(sp + (size_t)32 * DD, 0, 1);
#pragma unroll
    for (int j = 0; j < 4; ++j) {
      ldsBT[dseg + j][klrow]     = (_Float16)x0[j];
      ldsBT[dseg + 4 + j][klrow] = (_Float16)x1[j];
    }
    __syncthreads();

    H16 af[4], bf[2];
#pragma unroll
    for (int mf = 0; mf < 4; ++mf) {
      const int mq = (waveM << 6) + (mf << 4) + (lane & 15);
      const _Float16* ap = phiKT + ((size_t)b * NM + mq) * LK + kb + koff8;
      af[mf].h[0] = *(const v8h*)ap;
      af[mf].h[1] = *(const v8h*)(ap + 16);
    }
#pragma unroll
    for (int nf = 0; nf < 2; ++nf) {
      const int dl = (waveN << 5) + (nf << 4) + (lane & 15);
      bf[nf].h[0] = *(const v8h*)&ldsBT[dl][koff8];
      bf[nf].h[1] = *(const v8h*)&ldsBT[dl][koff8 + 16];
    }
#pragma unroll
    for (int mf = 0; mf < 4; ++mf)
#pragma unroll
      for (int nf = 0; nf < 2; ++nf)
        acc[mf][nf] = wmma16(af[mf].v, bf[nf].v, acc[mf][nf]);
    __syncthreads();
  }

#pragma unroll
  for (int mf = 0; mf < 4; ++mf) {
    const int mqb = (waveM << 6) + (mf << 4) + ((lane >> 4) << 3);
#pragma unroll
    for (int nf = 0; nf < 2; ++nf) {
      const int dg = dT + (waveN << 5) + (nf << 4) + (lane & 15);
      float* dst = S32p + (((size_t)(b * 4 + ks) * DD + dg) * NM + mqb);
      *(v8f*)dst = acc[mf][nf];   // S^T partial: [d][mq], 32B contiguous
    }
  }
}

// ---------------------------------------------------------------------------
// Kernel 5: reduce 4 K-split partials -> S16T [b][d][mq] f16
// ---------------------------------------------------------------------------
__global__ __launch_bounds__(256) void k_sred(const float* __restrict__ S32p,
                                              _Float16* __restrict__ S16T) {
  const int i = blockIdx.x * 256 + threadIdx.x;   // 262144 threads, 4 elems each
  const size_t base = (size_t)i << 2;
  const int b = (int)(base / (DD * NM));
  const size_t r = base % (DD * NM);
  const float* p = S32p + (size_t)b * 4 * DD * NM + r;
  v4f s = *(const v4f*)p;
  s = s + *(const v4f*)(p + (size_t)DD * NM);
  s = s + *(const v4f*)(p + (size_t)2 * DD * NM);
  s = s + *(const v4f*)(p + (size_t)3 * DD * NM);
  v4h h;
#pragma unroll
  for (int e = 0; e < 4; ++e) h[e] = (_Float16)s[e];
  *(v4h*)(S16T + base) = h;
}

// ---------------------------------------------------------------------------
// Kernel 6: den[b][l] = phi_q[l,:] . Z[b,:] + EPS
// ---------------------------------------------------------------------------
__global__ __launch_bounds__(256) void k_den(const _Float16* __restrict__ phiQ,
                                             const float* __restrict__ Z,
                                             float* __restrict__ den) {
  const int t = blockIdx.x * 256 + threadIdx.x;   // 16384
  const int b = t >> 11;
  const int l = t & 2047;
  const _Float16* p = phiQ + ((size_t)b * LQ + l) * NM;
  const float* zb = Z + b * NM;
  float s = 0.f;
  for (int j = 0; j < NM; j += 8) {
    v8h h = *(const v8h*)(p + j);
#pragma unroll
    for (int e = 0; e < 8; ++e) s += (float)h[e] * zb[j + e];
  }
  den[t] = s + 1e-8f;
}

// ---------------------------------------------------------------------------
// Kernel 7: num-GEMM (transposed: [d][l]) + fused epilogue.
// A = S16T rows, B = phiQ rows (both f16, direct global, L2-resident).
// out[l][d]: amp = content + num/den ; f_a = num/den. 32B contiguous stores.
// ---------------------------------------------------------------------------
__global__ __launch_bounds__(256) void k_num(const _Float16* __restrict__ S16T,
                                             const _Float16* __restrict__ phiQ,
                                             const float* __restrict__ denArr,
                                             const float* __restrict__ content,
                                             float* __restrict__ out) {
  const int tid = threadIdx.x, lane = tid & 31, wave = tid >> 5;
  const int waveM = wave & 1;   // 2 groups x 64 d
  const int waveN = wave >> 1;  // 4 groups x 32 l
  const int b = blockIdx.x >> 6;
  const int rest = blockIdx.x & 63;
  const int dT = (rest >> 4) << 7;   // 4 d-tiles of 128
  const int lT = (rest & 15) << 7;   // 16 l-tiles of 128

  v8f acc[4][2];
  const v8f vzero = {0.f, 0.f, 0.f, 0.f, 0.f, 0.f, 0.f, 0.f};
#pragma unroll
  for (int i = 0; i < 4; ++i)
#pragma unroll
    for (int j = 0; j < 2; ++j) acc[i][j] = vzero;

  const int koff8  = (lane >> 4) << 3;
  const int koff16 = (lane >> 4) << 4;

#pragma unroll
  for (int kt = 0; kt < 8; ++kt) {
    const int kb = kt << 5;
    H16 af[4], bf[2];
#pragma unroll
    for (int mf = 0; mf < 4; ++mf) {
      const int dg = dT + (waveM << 6) + (mf << 4) + (lane & 15);
      const _Float16* ap = S16T + ((size_t)b * DD + dg) * NM + kb + koff8;
      af[mf].h[0] = *(const v8h*)ap;
      af[mf].h[1] = *(const v8h*)(ap + 16);
    }
#pragma unroll
    for (int nf = 0; nf < 2; ++nf) {
      const int lg = lT + (waveN << 5) + (nf << 4) + (lane & 15);
      const _Float16* bp = phiQ + ((size_t)b * LQ + lg) * NM + kb + koff8;
      bf[nf].h[0] = *(const v8h*)bp;
      bf[nf].h[1] = *(const v8h*)(bp + 16);
    }
#pragma unroll
    for (int mf = 0; mf < 4; ++mf)
#pragma unroll
      for (int nf = 0; nf < 2; ++nf)
        acc[mf][nf] = wmma16(af[mf].v, bf[nf].v, acc[mf][nf]);
  }

  const size_t TOT = (size_t)NB * LQ * DD;
#pragma unroll
  for (int nf = 0; nf < 2; ++nf) {
    const int lg = lT + (waveN << 5) + (nf << 4) + (lane & 15);
    const float rdv = 1.0f / denArr[b * LQ + lg];
#pragma unroll
    for (int mf = 0; mf < 4; ++mf) {
      const int db = dT + (waveM << 6) + (mf << 4) + ((lane >> 4) << 3);
      const size_t idx = ((size_t)b * LQ + lg) * DD + db;
      v8f c0 = *(const v8f*)(content + idx);
      v8f fa, amp;
#pragma unroll
      for (int r = 0; r < 8; ++r) {
        fa[r] = acc[mf][nf][r] * rdv;
        amp[r] = c0[r] + fa[r];
      }
      *(v8f*)(out + idx) = amp;         // amp_updated
      *(v8f*)(out + TOT + idx) = fa;    // f_a
    }
  }
}

// ---------------------------------------------------------------------------
// Host launcher. Workspace layout (bytes):
//   W16T   @ 0          (256*512*2        =    262,144)
//   phiQ   @ 262144     (8*2048*256*2     =  8,388,608)
//   phiKT  @ 8650752    (8*256*8192*2     = 33,554,432)
//   S32p   @ 42205184   (8*4*512*256*4    = 16,777,216)
//   S16T   @ 58982400   (8*512*256*2      =  2,097,152)
//   Z      @ 61079552   (8*256*4          =      8,192)
//   den    @ 61087744   (8*2048*4         =     65,536)
//   total ~= 58.3 MB
// ---------------------------------------------------------------------------
extern "C" void kernel_launch(void* const* d_in, const int* in_sizes, int n_in,
                              void* d_out, int out_size, void* d_ws, size_t ws_size,
                              hipStream_t stream) {
  (void)in_sizes; (void)n_in; (void)out_size; (void)ws_size;
  const float* content = (const float*)d_in[0];
  const float* style   = (const float*)d_in[1];
  const float* W       = (const float*)d_in[2];

  char* wsb = (char*)d_ws;
  _Float16* W16T  = (_Float16*)(wsb + 0);
  _Float16* phiQ  = (_Float16*)(wsb + 262144);
  _Float16* phiKT = (_Float16*)(wsb + 8650752);
  float*    S32p  = (float*)(wsb + 42205184);
  _Float16* S16T  = (_Float16*)(wsb + 58982400);
  float*    Z     = (float*)(wsb + 61079552);
  float*    den   = (float*)(wsb + 61087744);
  float*    out   = (float*)d_out;

  k_wcvt <<<512,  256, 0, stream>>>(W, W16T);
  k_phik <<<1024, 256, 0, stream>>>(style, W16T, phiKT);
  k_phiq <<<256,  256, 0, stream>>>(content, W16T, phiQ);
  k_zsum <<<256,  256, 0, stream>>>(phiKT, Z);
  k_sgemm<<<256,  256, 0, stream>>>(style, phiKT, S32p);
  k_sred <<<1024, 256, 0, stream>>>(S32p, S16T);
  k_den  <<<64,   256, 0, stream>>>(phiQ, Z, den);
  k_num  <<<512,  256, 0, stream>>>(S16T, phiQ, den, content, out);
}
